// Region_AdapCNN_Block_28905129902573
// MI455X (gfx1250) — matrix operand order repaired and checked by reading, not compiled, not run on hardware
//
#include <hip/hip_runtime.h>
#include <hip/hip_bf16.h>

typedef __attribute__((ext_vector_type(16))) _Float16 v16h;
typedef __attribute__((ext_vector_type(8)))  _Float16 v8h;
typedef __attribute__((ext_vector_type(2)))  _Float16 v2h;
typedef __attribute__((ext_vector_type(8)))  float    v8f;

// ---------------------------------------------------------------------------
// WMMA helpers (CDNA5 wave32, D = A(16x32 f16) * B(32x16 f16) + C(16x16 f32))
// ---------------------------------------------------------------------------
__device__ __forceinline__ v8f wmma16(v16h a, v16h b, v8f c) {
  return __builtin_amdgcn_wmma_f32_16x16x32_f16(
      /*neg_a=*/false, a, /*neg_b=*/false, b,
      /*c_mod=*/(short)0, c, /*reuse_a=*/false, /*reuse_b=*/false);
}

__device__ __forceinline__ v16h cat8(v8h lo, v8h hi) {
  return __builtin_shufflevector(lo, hi, 0, 1, 2, 3, 4, 5, 6, 7,
                                 8, 9, 10, 11, 12, 13, 14, 15);
}

// A fragment (ISA 16-bit A layout): lane holds M=lane&15; halves 0..7 are
// K=kh+0..7, halves 8..15 are K=16+kh+0..7, kh=8*(lane>=16).
// p must point at (row=lane&15, k0+kh); two aligned 16B loads.
__device__ __forceinline__ v16h load_a_frag(const _Float16* p) {
  return cat8(*(const v8h*)p, *(const v8h*)(p + 16));
}

// B fragment from *transposed* storage Bt[n][k] (ldk halves per n-row):
// lane holds column n=lane&15; halves 0..7 are K=k0+kh+0..7, halves 8..15
// are K=k0+16+kh+0..7, kh=8*(lane>=16) (same order as A fragment).
__device__ __forceinline__ v16h load_b_fragT(const _Float16* Bt, int ldk,
                                             int k0, int col_base) {
  int lane = threadIdx.x & 31;
  const _Float16* p = Bt + (size_t)(col_base + (lane & 15)) * ldk + k0
                         + ((lane >> 4) << 3);
  return cat8(*(const v8h*)p, *(const v8h*)(p + 16));   // 2x ds_load_b128
}

// C/D layout: acc[r] <-> (M = r + 8*(lane>=16), N = lane&15)

// ---------------------------------------------------------------------------
// K0: feature f32 NCHW -> (a) d_out channels 0..63 (concat part 1)
//                        (b) feat16 NHWC f16 (WMMA A operand storage)
// LDS-tiled transpose: 64 ch x 64 x per block; padded rows vs bank conflicts.
// Grid: dim3(4 xblocks, 256 y, 4 b), 256 threads.
// ---------------------------------------------------------------------------
__global__ void prep_kernel(const float* __restrict__ feature,
                            float* __restrict__ out,
                            _Float16* __restrict__ feat16) {
  __shared__ alignas(32) float ts[64 * 65];
  int b = blockIdx.z, y = blockIdx.y, x0 = blockIdx.x * 64;
  int tid = threadIdx.x;
  for (int i = tid; i < 4096; i += 256) {        // coalesced along x
    int c = i >> 6, xx = i & 63;
    float v = feature[(((size_t)b * 64 + c) * 256 + y) * 256 + x0 + xx];
    ts[c * 65 + xx] = v;
    out[(((size_t)b * 96 + c) * 256 + y) * 256 + x0 + xx] = v;
  }
  __syncthreads();
  int pixel = tid >> 2, cg = (tid & 3) * 16;     // 64 pixels x 4 chan-groups
  v16h h;
#pragma unroll
  for (int k = 0; k < 16; ++k) h[k] = (_Float16)ts[(cg + k) * 65 + pixel];
  *(v16h*)(feat16 + (((size_t)b * 256 + y) * 256 + x0 + pixel) * 64 + cg) = h;
}

// ---------------------------------------------------------------------------
// K1: pack rw1 (co,ci,ky,kx) f32 -> w1p [tap][co][ci] f16 (K-contiguous)
// ---------------------------------------------------------------------------
__global__ void packw1_kernel(const float* __restrict__ rw1,
                              _Float16* __restrict__ w1p) {
  int idx = blockIdx.x * 256 + threadIdx.x;
  if (idx >= 64 * 64 * 9) return;
  int co = idx / 576, r = idx % 576, ci = r / 9, tap = r % 9;
  w1p[(tap * 64 + co) * 64 + ci] = (_Float16)rw1[idx];
}

// ---------------------------------------------------------------------------
// K2: conv1 64->64 3x3 pad1 + LeakyReLU(0.2), implicit GEMM per tap.
// Block: 256 thr (8 waves), 8x8 spatial patch (M=64) x 64 out ch (N=64).
// 10x10x64 haloed input patch staged once in LDS (zero-filled border);
// per-tap weights staged via uint4 copies. All fragments = b128 LDS loads.
// ---------------------------------------------------------------------------
__global__ void conv1_kernel(const _Float16* __restrict__ feat16,
                             const _Float16* __restrict__ w1p,
                             const float* __restrict__ rb1,
                             _Float16* __restrict__ h16) {
  __shared__ alignas(32) _Float16 wsm[64 * 64];       // [co][ci], 8 KB
  __shared__ alignas(32) _Float16 patch[10 * 10 * 64]; // 12.8 KB
  int b  = blockIdx.z;
  int by = blockIdx.y * 8, bx = blockIdx.x * 8;
  int tid = threadIdx.x, wave = tid >> 5, lane = tid & 31;
  int mt = wave >> 1, ntb = (wave & 1) * 2;

  for (int i = tid; i < 400; i += 256) {              // 100 pixels x 4 qtrs
    int pix = i >> 2, q = (i & 3) * 16;
    int py = pix / 10, px = pix % 10;
    int gy = by - 1 + py, gx = bx - 1 + px;
    v16h v = {};
    if (((unsigned)gy < 256u) && ((unsigned)gx < 256u))
      v = *(const v16h*)(feat16 + (((size_t)b * 256 + gy) * 256 + gx) * 64 + q);
    *(v16h*)(patch + pix * 64 + q) = v;
  }

  v8f acc[2] = {};
  int m  = mt * 16 + (lane & 15);
  int oy = m >> 3, ox = m & 7;
  int kh = (lane >> 4) << 3;

  for (int tap = 0; tap < 9; ++tap) {
    __syncthreads();
    {
      const uint4* src = (const uint4*)(w1p + tap * 4096);
      uint4* dst = (uint4*)wsm;
      for (int i = tid; i < 512; i += 256) dst[i] = src[i];
    }
    __syncthreads();
    int ky = tap / 3, kx = tap % 3;
    const _Float16* ap0 = patch + (((oy + ky) * 10) + ox + kx) * 64 + kh;
#pragma unroll
    for (int ks = 0; ks < 2; ++ks) {
      v16h a = load_a_frag(ap0 + ks * 32);
      v16h b0 = load_b_fragT(wsm, 64, ks * 32, ntb * 16);
      acc[0] = wmma16(a, b0, acc[0]);
      v16h b1 = load_b_fragT(wsm, 64, ks * 32, (ntb + 1) * 16);
      acc[1] = wmma16(a, b1, acc[1]);
    }
  }
  int mrow0 = mt * 16 + ((lane >> 4) << 3);
  int nc = lane & 15;
#pragma unroll
  for (int t = 0; t < 2; ++t) {
    int co = (ntb + t) * 16 + nc;
    float bv = rb1[co];
#pragma unroll
    for (int r = 0; r < 8; ++r) {
      int mm = mrow0 + r;
      int y = by + (mm >> 3), x = bx + (mm & 7);
      float v = acc[t][r] + bv;
      v = v >= 0.f ? v : 0.2f * v;                    // LeakyReLU(0.2)
      h16[(((size_t)b * 256 + y) * 256 + x) * 64 + co] = (_Float16)v;
    }
  }
}

// ---------------------------------------------------------------------------
// K3: conv2 64->1 3x3 pad1 + tanh. One pixel per thread, vectorized K loads.
// ---------------------------------------------------------------------------
__global__ void conv2_kernel(const _Float16* __restrict__ h16,
                             const float* __restrict__ rw2,
                             const float* __restrict__ rb2,
                             float* __restrict__ adapt) {
  __shared__ float w[576];
  int tid = threadIdx.x;
  for (int i = tid; i < 576; i += 256) w[i] = rw2[i];
  __syncthreads();
  int idx = blockIdx.x * 256 + tid;                   // 4*256*256
  int b = idx >> 16, r = idx & 65535, y = r >> 8, x = r & 255;
  float acc = rb2[0];
  for (int ky = 0; ky < 3; ++ky) {
    int yy = y + ky - 1; if ((unsigned)yy >= 256u) continue;
    for (int kx = 0; kx < 3; ++kx) {
      int xx = x + kx - 1; if ((unsigned)xx >= 256u) continue;
      const _Float16* hp = h16 + (((size_t)b * 256 + yy) * 256 + xx) * 64;
      const float* wp = w + (ky * 3 + kx);            // rw2[ci][ky][kx], str 9
      float s = 0.f;
#pragma unroll
      for (int cg = 0; cg < 8; ++cg) {
        v8h hv = *(const v8h*)(hp + cg * 8);
#pragma unroll
        for (int j = 0; j < 8; ++j) s += (float)hv[j] * wp[(cg * 8 + j) * 9];
      }
      acc += s;
    }
  }
  adapt[idx] = tanhf(acc);
}

// ---------------------------------------------------------------------------
// K4: adaptive-avg gram per tile: 66x66 (haloed) -> 32x32, write f16 A-matrix.
// ---------------------------------------------------------------------------
__global__ void pool_kernel(const float* __restrict__ adapt,
                            _Float16* __restrict__ gramA) {
  __shared__ float tmp[32 * 66];
  int bt = blockIdx.x;
  int b = bt >> 4, t = bt & 15, ti = t >> 2, tj = t & 3;
  int tid = threadIdx.x;
  for (int i = tid; i < 32 * 66; i += 256) {
    int p = i / 66, w = i % 66;
    int s = p * 66 / 32, e = ((p + 1) * 66 + 31) / 32;
    int gw = tj * 64 - 1 + w;
    float sum = 0.f;
    if ((unsigned)gw < 256u) {
      for (int hh = s; hh < e; ++hh) {
        int gh = ti * 64 - 1 + hh;
        if ((unsigned)gh < 256u) sum += adapt[((size_t)b * 256 + gh) * 256 + gw];
      }
    }
    tmp[i] = sum / (float)(e - s);
  }
  __syncthreads();
  for (int i = tid; i < 1024; i += 256) {
    int p = i >> 5, q = i & 31;
    int s = q * 66 / 32, e = ((q + 1) * 66 + 31) / 32;
    float sum = 0.f;
    for (int w = s; w < e; ++w) sum += tmp[p * 66 + w];
    gramA[(size_t)bt * 1024 + i] = (_Float16)(sum / (float)(e - s));
  }
}

// ---------------------------------------------------------------------------
// K5/K6/K7: M=64 GEMM: C(64xN) = A(64xK f16) @ B(KxN f32).
// B chunk (32x64) converted f32->f16 into *transposed* LDS [n][k] so B
// fragments are contiguous 32B ds_load_b128 pairs. 128 thr (4 waves).
// MODE 0: +bias, ReLU, store f16 row-major.
// MODE 1: +bias, scatter dyn weights [bt][tap][co][ci] f16, dyn bias f32.
// ---------------------------------------------------------------------------
template <int MODE>
__global__ void gemm64_kernel(const _Float16* __restrict__ A, int K,
                              const float* __restrict__ B, int N,
                              const float* __restrict__ bias,
                              _Float16* __restrict__ C16,
                              float* __restrict__ dynb) {
  __shared__ alignas(32) _Float16 bsT[64 * 32];       // [n][k], 4 KB
  int nb = blockIdx.x * 64;
  int tid = threadIdx.x, wave = tid >> 5, lane = tid & 31;
  v8f acc[4] = {};
  for (int k0 = 0; k0 < K; k0 += 32) {
    __syncthreads();
    for (int i = tid; i < 1024; i += 128) {           // coalesced along N
      int nn = i & 63, kk = (i >> 6) * 2;
      int col = nb + nn;
      float v0 = 0.f, v1 = 0.f;
      if (col < N) {
        v0 = B[(size_t)(k0 + kk) * N + col];
        v1 = B[(size_t)(k0 + kk + 1) * N + col];
      }
      v2h pk; pk[0] = (_Float16)v0; pk[1] = (_Float16)v1;
      *(v2h*)(bsT + nn * 32 + kk) = pk;               // ds_store_b32
    }
    __syncthreads();
    v16h a = load_a_frag(A + (size_t)(wave * 16 + (lane & 15)) * K + k0
                           + ((lane >> 4) << 3));
#pragma unroll
    for (int nt = 0; nt < 4; ++nt) {
      v16h bb = load_b_fragT(bsT, 32, 0, nt * 16);
      acc[nt] = wmma16(a, bb, acc[nt]);
    }
  }
  int mrow0 = wave * 16 + ((lane >> 4) << 3);
  int ncol = lane & 15;
#pragma unroll
  for (int nt = 0; nt < 4; ++nt) {
#pragma unroll
    for (int r = 0; r < 8; ++r) {
      int row = mrow0 + r;
      int col = nb + nt * 16 + ncol;
      if (col >= N) continue;
      float v = acc[nt][r] + bias[col];
      if (MODE == 0) {
        v = v > 0.f ? v : 0.f;
        C16[(size_t)row * N + col] = (_Float16)v;
      } else {
        if (col < 18432) {                            // dyn conv weights
          int co = col / 576, rr = col % 576, ci = rr / 9, tap = rr % 9;
          C16[(((size_t)row * 9 + tap) * 32 + co) * 64 + ci] = (_Float16)v;
        } else {                                      // dyn conv bias
          dynb[row * 32 + (col - 18432)] = v;
        }
      }
    }
  }
}

// ---------------------------------------------------------------------------
// K8: per-tile dynamic conv (64->32, 3x3 VALID on haloed 66x66 -> 64x64).
// Block: 256 thr (8 waves), 8x8 output patch (M=64) x 32 ch (N=32).
// Tile weights [tap][co][ci] (36 KB) + 10x10x64 input patch staged in LDS.
// ---------------------------------------------------------------------------
__global__ void dynconv_kernel(const _Float16* __restrict__ feat16,
                               const _Float16* __restrict__ dynw,
                               const float* __restrict__ dynb,
                               float* __restrict__ out) {
  __shared__ alignas(32) _Float16 wsm[9 * 32 * 64];     // 36864 B
  __shared__ alignas(32) _Float16 patch[10 * 10 * 64];  // 12.8 KB
  int bt = blockIdx.z;
  int b = bt >> 4, t = bt & 15, ti = t >> 2, tj = t & 3;
  int sby = blockIdx.y * 8, sbx = blockIdx.x * 8;
  int tid = threadIdx.x;
  {
    const uint4* src = (const uint4*)(dynw + (size_t)bt * 9 * 32 * 64);
    uint4* dst = (uint4*)wsm;
    for (int i = tid; i < 1152; i += 256) dst[i] = src[i];
  }
  int gy0 = ti * 64 - 1 + sby, gx0 = tj * 64 - 1 + sbx;
  for (int i = tid; i < 400; i += 256) {
    int pix = i >> 2, q = (i & 3) * 16;
    int py = pix / 10, px = pix % 10;
    int gy = gy0 + py, gx = gx0 + px;
    v16h v = {};
    if (((unsigned)gy < 256u) && ((unsigned)gx < 256u))
      v = *(const v16h*)(feat16 + (((size_t)b * 256 + gy) * 256 + gx) * 64 + q);
    *(v16h*)(patch + pix * 64 + q) = v;
  }
  __syncthreads();

  int wave = tid >> 5, lane = tid & 31;
  int mt = wave >> 1, nt = wave & 1;
  v8f acc = {};
  int m  = mt * 16 + (lane & 15);
  int oy = m >> 3, ox = m & 7;                        // local output pos
  int kh = (lane >> 4) << 3;

  for (int tap = 0; tap < 9; ++tap) {
    int ky = tap / 3, kx = tap % 3;
    const _Float16* ap0 = patch + (((oy + ky) * 10) + ox + kx) * 64 + kh;
    const _Float16* bp = wsm + tap * 32 * 64;         // [co][ci]
#pragma unroll
    for (int ks = 0; ks < 2; ++ks) {
      v16h a = load_a_frag(ap0 + ks * 32);
      v16h bb = load_b_fragT(bp, 64, ks * 32, nt * 16);
      acc = wmma16(a, bb, acc);
    }
  }
  int mrow0 = mt * 16 + ((lane >> 4) << 3);
  int co = nt * 16 + (lane & 15);
  float bv = dynb[bt * 32 + co];
#pragma unroll
  for (int r = 0; r < 8; ++r) {
    int mm = mrow0 + r;
    int gy = ti * 64 + sby + (mm >> 3);
    int gx = tj * 64 + sbx + (mm & 7);
    out[(((size_t)b * 96 + 64 + co) * 256 + gy) * 256 + gx] = acc[r] + bv;
  }
}

// ---------------------------------------------------------------------------
// Workspace layout (bytes, 256-aligned), total ~71 MB
// ---------------------------------------------------------------------------
#define WS_FEAT16 ((size_t)0)          // 33,554,432  f16 NHWC feature
#define WS_H16    ((size_t)33554432)   // 33,554,432  f16 NHWC conv1 output
#define WS_W1P    ((size_t)67108864)   //     73,728  f16 packed rw1 [tap][co][ci]
#define WS_ADAPT  ((size_t)67182592)   //  1,048,576  f32 tanh map
#define WS_GRAM   ((size_t)68231168)   //    131,072  f16 gram A (64x1024)
#define WS_G1     ((size_t)68362240)   //    262,144  f16 g1 (64x2048)
#define WS_G2     ((size_t)68624384)   //    262,144  f16 g2 (64x2048)
#define WS_DYNW   ((size_t)68886528)   //  2,359,296  f16 dyn weights [bt][tap][co][ci]
#define WS_DYNB   ((size_t)71245824)   //      8,192  f32 dyn bias

extern "C" void kernel_launch(void* const* d_in, const int* in_sizes, int n_in,
                              void* d_out, int out_size, void* d_ws, size_t ws_size,
                              hipStream_t stream) {
  (void)in_sizes; (void)n_in; (void)out_size; (void)ws_size;
  const float* feature = (const float*)d_in[0];
  const float* rw1 = (const float*)d_in[1];
  const float* rb1 = (const float*)d_in[2];
  const float* rw2 = (const float*)d_in[3];
  const float* rb2 = (const float*)d_in[4];
  const float* fw1 = (const float*)d_in[5];
  const float* fb1 = (const float*)d_in[6];
  const float* fw2 = (const float*)d_in[7];
  const float* fb2 = (const float*)d_in[8];
  const float* fw3 = (const float*)d_in[9];
  const float* fb3 = (const float*)d_in[10];
  float* out = (float*)d_out;
  char* ws = (char*)d_ws;

  _Float16* feat16 = (_Float16*)(ws + WS_FEAT16);
  _Float16* h16    = (_Float16*)(ws + WS_H16);
  _Float16* w1p    = (_Float16*)(ws + WS_W1P);
  float*    adapt  = (float*)   (ws + WS_ADAPT);
  _Float16* gramA  = (_Float16*)(ws + WS_GRAM);
  _Float16* g1     = (_Float16*)(ws + WS_G1);
  _Float16* g2     = (_Float16*)(ws + WS_G2);
  _Float16* dynw   = (_Float16*)(ws + WS_DYNW);
  float*    dynb   = (float*)   (ws + WS_DYNB);

  prep_kernel<<<dim3(4, 256, 4), 256, 0, stream>>>(feature, out, feat16);
  packw1_kernel<<<144, 256, 0, stream>>>(rw1, w1p);
  conv1_kernel<<<dim3(32, 32, 4), 256, 0, stream>>>(feat16, w1p, rb1, h16);
  conv2_kernel<<<1024, 256, 0, stream>>>(h16, rw2, rb2, adapt);
  pool_kernel<<<64, 256, 0, stream>>>(adapt, gramA);
  gemm64_kernel<0><<<32, 128, 0, stream>>>(gramA, 1024, fw1, 2048, fb1, g1, nullptr);
  gemm64_kernel<0><<<32, 128, 0, stream>>>(g1, 2048, fw2, 2048, fb2, g2, nullptr);
  gemm64_kernel<1><<<289, 128, 0, stream>>>(g2, 2048, fw3, 18464, fb3, dynw, dynb);
  dynconv_kernel<<<dim3(8, 8, 64), 256, 0, stream>>>(feat16, dynw, dynb, out);
}
